// Seasonal_18786186953497
// MI455X (gfx1250) — matrix operand order repaired and checked by reading, not compiled, not run on hardware
//
#include <hip/hip_runtime.h>
#include <hip/hip_bf16.h>
#include <math.h>

typedef __bf16 bf16_t;
typedef __attribute__((ext_vector_type(16))) __bf16 v16bf;
typedef __attribute__((ext_vector_type(8)))  __bf16 v8bf;
typedef __attribute__((ext_vector_type(8)))  float  v8f;

#define NN 4096
#define BB 2048

union FragB16 { v16bf v; v8bf h[2]; };

// ---------------------------------------------------------------------------
// Kernel 1: h[m] = (1/N) * sum_k d[k] * cos(2*pi*k*m/N), split into bf16 hi/lo.
// One block per m; 256-thread LDS reduction. Angle computed from (k*m mod N)
// so cosf args stay in [0, 2*pi) (no large-argument range-reduction error).
// ---------------------------------------------------------------------------
__global__ void build_h_kernel(const float* __restrict__ diag,
                               bf16_t* __restrict__ h_hi,
                               bf16_t* __restrict__ h_lo) {
  __shared__ float red[256];
  const int m = blockIdx.x;
  const int t = threadIdx.x;
  const float w = 6.283185307179586f / (float)NN;
  float s = 0.0f;
  for (int k = t; k < NN; k += 256) {
    int r = (k * m) & (NN - 1);
    s += diag[k] * cosf(w * (float)r);
  }
  red[t] = s;
  __syncthreads();
  for (int off = 128; off > 0; off >>= 1) {
    if (t < off) red[t] += red[t + off];
    __syncthreads();
  }
  if (t == 0) {
    float h = red[0] * (1.0f / (float)NN);
    bf16_t hi = (bf16_t)h;
    h_hi[m] = hi;
    h_lo[m] = (bf16_t)(h - (float)hi);
  }
}

// ---------------------------------------------------------------------------
// Kernel 2: split x (f32) into bf16 hi/lo pair. 8 elements per thread.
// ---------------------------------------------------------------------------
__global__ void split_x_kernel(const float* __restrict__ xin,
                               bf16_t* __restrict__ xh,
                               bf16_t* __restrict__ xl) {
  size_t i0 = ((size_t)blockIdx.x * 256 + threadIdx.x) * 8;
  v8f v = *(const v8f*)(xin + i0);
  v8bf hi, lo;
#pragma unroll
  for (int e = 0; e < 8; ++e) {
    float f = v[e];
    bf16_t h = (bf16_t)f;
    hi[e] = h;
    lo[e] = (bf16_t)(f - (float)h);
  }
  *(v8bf*)(xh + i0) = hi;
  *(v8bf*)(xl + i0) = lo;
}

// ---------------------------------------------------------------------------
// Kernel 3: materialize C^T in bf16 hi/lo:  Ct[n*N + k] = h[(k - n) mod N].
// Row-major N x K layout makes WMMA B-fragment loads contiguous (b128).
// ---------------------------------------------------------------------------
__global__ void build_ct_kernel(const bf16_t* __restrict__ h_hi,
                                const bf16_t* __restrict__ h_lo,
                                bf16_t* __restrict__ cth,
                                bf16_t* __restrict__ ctl) {
  size_t i0 = ((size_t)blockIdx.x * 256 + threadIdx.x) * 8;
  int n  = (int)(i0 >> 12);
  int k0 = (int)(i0 & (NN - 1));
  v8bf vh, vl;
#pragma unroll
  for (int e = 0; e < 8; ++e) {
    int mm = (k0 + e - n) & (NN - 1);
    vh[e] = h_hi[mm];
    vl[e] = h_lo[mm];
  }
  *(v8bf*)(cth + i0) = vh;
  *(v8bf*)(ctl + i0) = vl;
}

// ---------------------------------------------------------------------------
// Kernel 4: y = tanh(x @ C) via bf16x3-split WMMA (f32 accumulate).
// 8 waves per block; each wave owns a 64x64 output block (4x4 WMMA tiles).
// Per K-step of 32: 48 v_wmma_f32_16x16x32_bf16 (hi*hi + lo*hi + hi*lo).
// ---------------------------------------------------------------------------
__global__ __launch_bounds__(256, 1) void gemm_circ_tanh_kernel(
    const bf16_t* __restrict__ xh,  const bf16_t* __restrict__ xl,
    const bf16_t* __restrict__ cth, const bf16_t* __restrict__ ctl,
    float* __restrict__ out) {
  const int lane = threadIdx.x & 31;
  const int wave = threadIdx.x >> 5;       // 0..7
  const int wm = wave & 1;                 // 2 wave-rows
  const int wn = wave >> 1;                // 4 wave-cols
  const int m_base = blockIdx.x * 128 + wm * 64;
  const int n_base = blockIdx.y * 256 + wn * 64;
  const int lr = lane & 15;                // row-within-tile selector
  const int hl = lane >> 4;                // lane half

  const v8f vzero = {0.f, 0.f, 0.f, 0.f, 0.f, 0.f, 0.f, 0.f};
  v8f acc[4][4];
#pragma unroll
  for (int i = 0; i < 4; ++i)
#pragma unroll
    for (int j = 0; j < 4; ++j) acc[i][j] = vzero;

  size_t aoff[4], boff[4];
#pragma unroll
  for (int i = 0; i < 4; ++i) aoff[i] = (size_t)(m_base + i * 16 + lr) * NN;
#pragma unroll
  for (int j = 0; j < 4; ++j) boff[j] = (size_t)(n_base + j * 16 + lr) * NN;

  for (int kb = 0; kb < NN; kb += 32) {
    const int ka0 = kb + hl * 8;           // A frag: K = ka0..ka0+7 (elems 0..7)
    const int ka1 = kb + 16 + hl * 8;      //         K = ka1..ka1+7 (elems 8..15)
    const int kb0 = kb + hl * 8;           // B frag: K = kb0..kb0+7, kb0+16..kb0+23

    FragB16 ah[4], al[4];
#pragma unroll
    for (int i = 0; i < 4; ++i) {
      ah[i].h[0] = *(const v8bf*)(xh + aoff[i] + ka0);
      ah[i].h[1] = *(const v8bf*)(xh + aoff[i] + ka1);
      al[i].h[0] = *(const v8bf*)(xl + aoff[i] + ka0);
      al[i].h[1] = *(const v8bf*)(xl + aoff[i] + ka1);
    }
#pragma unroll
    for (int j = 0; j < 4; ++j) {
      FragB16 bh, bl;
      bh.h[0] = *(const v8bf*)(cth + boff[j] + kb0);
      bh.h[1] = *(const v8bf*)(cth + boff[j] + kb0 + 16);
      bl.h[0] = *(const v8bf*)(ctl + boff[j] + kb0);
      bl.h[1] = *(const v8bf*)(ctl + boff[j] + kb0 + 16);
#pragma unroll
      for (int i = 0; i < 4; ++i) {
        acc[i][j] = __builtin_amdgcn_wmma_f32_16x16x32_bf16(
            false, ah[i].v, false, bh.v, (short)0, acc[i][j], false, false);
        acc[i][j] = __builtin_amdgcn_wmma_f32_16x16x32_bf16(
            false, al[i].v, false, bh.v, (short)0, acc[i][j], false, false);
        acc[i][j] = __builtin_amdgcn_wmma_f32_16x16x32_bf16(
            false, ah[i].v, false, bl.v, (short)0, acc[i][j], false, false);
      }
    }
  }

  // Epilogue: tanh + store. C/D layout: col = lane%16, row = v + 8*(lane/16).
#pragma unroll
  for (int i = 0; i < 4; ++i) {
#pragma unroll
    for (int j = 0; j < 4; ++j) {
      const int ncol = n_base + j * 16 + lr;
#pragma unroll
      for (int v = 0; v < 8; ++v) {
        const int mrow = m_base + i * 16 + v + 8 * hl;
        out[(size_t)mrow * NN + ncol] = tanhf(acc[i][j][v]);
      }
    }
  }
}

// ---------------------------------------------------------------------------
// Host-side launcher. Inputs: d_in[0]=x (f32, 2048x4096), d_in[1]=diagonal
// (f32, 4096), d_in[2]=f_basis (complex64) -- UNUSED (closed-form circulant).
// Workspace: h_hi/h_lo (16 KB) | x_hi/x_lo (32 MB) | Ct_hi/Ct_lo (64 MB).
// ---------------------------------------------------------------------------
extern "C" void kernel_launch(void* const* d_in, const int* in_sizes, int n_in,
                              void* d_out, int out_size, void* d_ws, size_t ws_size,
                              hipStream_t stream) {
  (void)in_sizes; (void)n_in; (void)out_size; (void)ws_size;
  const float* x    = (const float*)d_in[0];
  const float* diag = (const float*)d_in[1];
  float* out = (float*)d_out;

  char* ws = (char*)d_ws;
  bf16_t* h_hi = (bf16_t*)(ws);
  bf16_t* h_lo = (bf16_t*)(ws + 8192);
  bf16_t* xh   = (bf16_t*)(ws + 16384);
  bf16_t* xl   = xh  + (size_t)BB * NN;
  bf16_t* cth  = xl  + (size_t)BB * NN;
  bf16_t* ctl  = cth + (size_t)NN * NN;

  build_h_kernel<<<NN, 256, 0, stream>>>(diag, h_hi, h_lo);
  split_x_kernel<<<(BB * NN) / (8 * 256), 256, 0, stream>>>(x, xh, xl);
  build_ct_kernel<<<(NN * (NN / (8 * 256))), 256, 0, stream>>>(h_hi, h_lo, cth, ctl);

  dim3 grid(BB / 128, NN / 256);
  gemm_circ_tanh_kernel<<<grid, 256, 0, stream>>>(xh, xl, cth, ctl, out);
}
